// MLP_61761629716703
// MI455X (gfx1250) — compile-verified, matrix operand order checked by reading, not run on hardware
//
#include <hip/hip_runtime.h>
#include <hip/hip_bf16.h>

typedef __attribute__((ext_vector_type(16))) _Float16 v16h;
typedef __attribute__((ext_vector_type(8)))  _Float16 v8h;
typedef __attribute__((ext_vector_type(8)))  float    v8f;
typedef __attribute__((ext_vector_type(4)))  float    v4f;
typedef unsigned int u32x4 __attribute__((ext_vector_type(4)));
typedef int          i32x4 __attribute__((ext_vector_type(4)));
typedef int          i32x8 __attribute__((ext_vector_type(8)));

#define DIM  1024
#define HID  4096
#define RANK 16

#if defined(__has_builtin)
#if __has_builtin(__builtin_amdgcn_tensor_load_to_lds)
#define USE_TDM 1
#endif
#endif
#ifndef USE_TDM
#define USE_TDM 0
#endif

// ---------------- fbv = fast_b @ v  (RANK x DIM) ----------------
__global__ void k_fbv(const float* __restrict__ fb, const float* __restrict__ v,
                      float* __restrict__ fbv) {
  int id = blockIdx.x * 256 + threadIdx.x;   // RANK*DIM total
  int r = id >> 10, j = id & (DIM - 1);
  float s = 0.f;
#pragma unroll
  for (int q = 0; q < RANK; q++) s += fb[r * RANK + q] * v[q * DIM + j];
  fbv[r * DIM + j] = s;
}

// ------- ternarize one row of W (+ optional folded low-rank adapter) -------
template <int COLS, bool ADAPT>
__global__ void k_prep_w(const float* __restrict__ W, const float* __restrict__ u,
                         const float* __restrict__ fbv, const float* __restrict__ gate_p,
                         _Float16* __restrict__ Wo) {
  const int i = blockIdx.x;      // output row
  const int tid = threadIdx.x;
  __shared__ float red[256];
  __shared__ float urow[RANK];
  if (ADAPT && tid < RANK) urow[tid] = u[i * RANK + tid];
  float s = 0.f;
  for (int j = tid; j < COLS; j += 256) s += fabsf(W[i * COLS + j]);
  red[tid] = s;
  __syncthreads();
  for (int off = 128; off > 0; off >>= 1) {
    if (tid < off) red[tid] += red[tid + off];
    __syncthreads();
  }
  float scale = fmaxf(red[0] * (1.0f / COLS), 1e-5f);  // clip(mean|w|, 1e-5)
  float g = ADAPT ? gate_p[0] : 0.f;
  for (int j = tid; j < COLS; j += 256) {
    float w = W[i * COLS + j];
    float t = fminf(fmaxf(rintf(w / scale), -1.f), 1.f);  // round-half-even like jnp.round
    float val = t * scale;
    if (ADAPT) {
      float a = 0.f;
#pragma unroll
      for (int r = 0; r < RANK; r++) a += urow[r] * fbv[r * DIM + j];
      val += g * a;
    }
    Wo[i * COLS + j] = (_Float16)val;
  }
}

// ---------------- f32 -> f16 convert (vectorized) ----------------
__global__ void k_cvt(const float* __restrict__ x, _Float16* __restrict__ xh, int n) {
  int base = (blockIdx.x * 256 + threadIdx.x) * 8;
  if (base >= n) return;
  v4f a = *(const v4f*)(x + base);
  v4f b = *(const v4f*)(x + base + 4);
  v8h o;
#pragma unroll
  for (int k = 0; k < 4; k++) { o[k] = (_Float16)a[k]; o[k + 4] = (_Float16)b[k]; }
  *(v8h*)(xh + base) = o;
}

#if USE_TDM
// Issue one TDM 2-D tile load: 128 rows x 32 halves from row-major (stride=kdim)
// global memory into LDS, with HW padding of 4 DWORDs (16B) after every
// 16 DWORDs (64B = one 32-half row)  =>  LDS row stride 80B (40 halves).
// D# packing per CDNA5 ISA 8.3/8.4 (group0 128b, group1 256b, groups2/3 zero).
__device__ __forceinline__ void tdm_load_tile(const _Float16* gsrc,
                                              unsigned lds_byte_off, int kdim) {
  unsigned long long ga = (unsigned long long)(size_t)gsrc;
  u32x4 g0 = { 1u,                                   // count=1, user-mode D#
               lds_byte_off,                         // lds_addr (bytes)
               (unsigned)(ga & 0xffffffffull),       // global_addr[31:0]
               (unsigned)((ga >> 32) & 0x01ffffffull) | (2u << 30) }; // [56:32] + type=2
  i32x8 g1 = { (int)((1u << 16) | (1u << 20) | (3u << 22) | (3u << 25)),
               //  data_size=2B | pad_enable | pad_interval=16DW | pad_amount=4DW
               (int)(((unsigned)kdim & 0xffffu) << 16),  // tensor_dim0[15:0] (=kdim)
               (int)(0x8000u << 16),                     // tensor_dim1[15:0] = 32768 rows
               (int)(32u << 16),                         // tile_dim0 = 32 elements
               128,                                      // tile_dim1 = 128 rows
               kdim,                                     // tensor_dim0_stride[31:0]
               0, 0 };
  i32x4 gz = { 0, 0, 0, 0 };
#if __clang_major__ >= 23
  i32x8 g4 = { 0, 0, 0, 0, 0, 0, 0, 0 };
  __builtin_amdgcn_tensor_load_to_lds(g0, g1, gz, gz, g4, 0);
#else
  __builtin_amdgcn_tensor_load_to_lds(g0, g1, gz, gz, 0);
#endif
}
#endif

// ---------------- tiled WMMA GEMM: C = A(MxK) * Bt(NxK)^T ----------------
// Block: 256 thr = 8 wave32; tile 128x128, K-step 32.
// Wave (wm 0..3, wn 0..1) owns 32x64: 2x4 accumulators of 16x16.
// TDM path: double-buffered LDS, DMA of next K-tile overlapped with WMMAs.
template <int NDIM, int KDIM, bool SQRELU>
__global__ __launch_bounds__(256) void k_gemm(const _Float16* __restrict__ A,
                                              const _Float16* __restrict__ Bt,
                                              _Float16* __restrict__ outh,
                                              float* __restrict__ outf, int M) {
  const int tid = threadIdx.x;
  const int bm = blockIdx.x * 128;
  const int bn = blockIdx.y * 128;
  const int wid = tid >> 5, lane = tid & 31;
  const int wm = wid & 3, wn = wid >> 2;
  const int lr = lane & 15, lh = lane >> 4;

  v8f zero{};
  v8f acc[2][4];
#pragma unroll
  for (int a = 0; a < 2; a++)
#pragma unroll
    for (int b = 0; b < 4; b++) acc[a][b] = zero;

  union F16x16 { v16h v; v8h h[2]; };

#if USE_TDM
  __shared__ _Float16 sA[2][128 * 40];   // 40-half rows (80B, TDM pad keeps 16B align)
  __shared__ _Float16 sB[2][128 * 40];
  const _Float16* gA = A + (size_t)bm * KDIM;
  const _Float16* gB = Bt + (size_t)bn * KDIM;
  if (tid < 32) {  // wave0 issues the DMA descriptors
    tdm_load_tile(gA, (unsigned)(size_t)&sA[0][0], KDIM);
    tdm_load_tile(gB, (unsigned)(size_t)&sB[0][0], KDIM);
  }
  __builtin_amdgcn_s_wait_tensorcnt(0);
  __syncthreads();
  int cur = 0;
  for (int k0 = 0; k0 < KDIM; k0 += 32) {
    if (k0 + 32 < KDIM && tid < 32) {  // kick off next tile into the other buffer
      tdm_load_tile(gA + k0 + 32, (unsigned)(size_t)&sA[cur ^ 1][0], KDIM);
      tdm_load_tile(gB + k0 + 32, (unsigned)(size_t)&sB[cur ^ 1][0], KDIM);
    }
    const _Float16* pA = sA[cur];
    const _Float16* pB = sB[cur];
    // ISA 16-bit A layout: lane (lh,lr) holds row lr, K = 8*lh..+7 then 16+8*lh..+7
    F16x16 fa[2], fb[4];
#pragma unroll
    for (int a = 0; a < 2; a++) {
      const _Float16* p = pA + (wm * 32 + a * 16 + lr) * 40 + 8 * lh;
      fa[a].h[0] = *(const v8h*)p;
      fa[a].h[1] = *(const v8h*)(p + 16);
    }
#pragma unroll
    for (int b = 0; b < 4; b++) {
      const _Float16* p = pB + (wn * 64 + b * 16 + lr) * 40 + 8 * lh;
      fb[b].h[0] = *(const v8h*)p;
      fb[b].h[1] = *(const v8h*)(p + 16);
    }
#pragma unroll
    for (int a = 0; a < 2; a++)
#pragma unroll
      for (int b = 0; b < 4; b++)
        acc[a][b] = __builtin_amdgcn_wmma_f32_16x16x32_f16(
            false, fa[a].v, false, fb[b].v, (short)0, acc[a][b], false, false);
    __builtin_amdgcn_s_wait_tensorcnt(0);  // next-tile DMA done (no-op for waves 1..7)
    __syncthreads();                       // everyone finished reading cur; flip
    cur ^= 1;
  }
#else
  __shared__ _Float16 sA[128 * 40];
  __shared__ _Float16 sB[128 * 40];
  for (int k0 = 0; k0 < KDIM; k0 += 32) {
#pragma unroll
    for (int i = 0; i < 2; i++) {
      int c = tid + 256 * i;          // 512 chunks of 8 halves
      int row = c >> 2;
      int kc = (c & 3) << 3;
      *(v8h*)(sA + row * 40 + kc) = *(const v8h*)(A + (size_t)(bm + row) * KDIM + k0 + kc);
      *(v8h*)(sB + row * 40 + kc) = *(const v8h*)(Bt + (size_t)(bn + row) * KDIM + k0 + kc);
    }
    __syncthreads();
    if (k0 + 32 < KDIM)
      __builtin_prefetch(A + (size_t)(bm + (tid >> 1)) * KDIM + k0 + 32, 0, 1);
    F16x16 fa[2], fb[4];
#pragma unroll
    for (int a = 0; a < 2; a++) {
      const _Float16* p = sA + (wm * 32 + a * 16 + lr) * 40 + 8 * lh;
      fa[a].h[0] = *(const v8h*)p;
      fa[a].h[1] = *(const v8h*)(p + 16);
    }
#pragma unroll
    for (int b = 0; b < 4; b++) {
      const _Float16* p = sB + (wn * 64 + b * 16 + lr) * 40 + 8 * lh;
      fb[b].h[0] = *(const v8h*)p;
      fb[b].h[1] = *(const v8h*)(p + 16);
    }
#pragma unroll
    for (int a = 0; a < 2; a++)
#pragma unroll
      for (int b = 0; b < 4; b++)
        acc[a][b] = __builtin_amdgcn_wmma_f32_16x16x32_f16(
            false, fa[a].v, false, fb[b].v, (short)0, acc[a][b], false, false);
    __syncthreads();
  }
#endif

  // C/D layout: VGPR r -> row (r + 8*lh), col lr within 16x16 tile
#pragma unroll
  for (int a = 0; a < 2; a++) {
#pragma unroll
    for (int b = 0; b < 4; b++) {
      int mb = bm + wm * 32 + a * 16 + lh * 8;
      int nb = bn + wn * 64 + b * 16 + lr;
#pragma unroll
      for (int r = 0; r < 8; r++) {
        float val = acc[a][b][r];
        size_t idx = (size_t)(mb + r) * NDIM + nb;
        if (SQRELU) {
          float rl = fmaxf(val, 0.f);
          outh[idx] = (_Float16)(rl * rl);
        } else {
          outf[idx] = val;
        }
      }
    }
  }
}

extern "C" void kernel_launch(void* const* d_in, const int* in_sizes, int n_in,
                              void* d_out, int out_size, void* d_ws, size_t ws_size,
                              hipStream_t stream) {
  const float* x      = (const float*)d_in[0];
  const float* fast_b = (const float*)d_in[1];
  const float* w_fc   = (const float*)d_in[2];
  const float* w_proj = (const float*)d_in[3];
  const float* u      = (const float*)d_in[4];
  const float* v      = (const float*)d_in[5];
  const float* gate   = (const float*)d_in[6];
  float* out = (float*)d_out;

  const int M = in_sizes[0] / DIM;  // 16384 tokens (multiple of 128)

  char* ws = (char*)d_ws;
  _Float16* xh = (_Float16*)ws; ws += (size_t)M * DIM * sizeof(_Float16);
  _Float16* W1 = (_Float16*)ws; ws += (size_t)HID * DIM * sizeof(_Float16);
  _Float16* W2 = (_Float16*)ws; ws += (size_t)DIM * HID * sizeof(_Float16);
  _Float16* C1 = (_Float16*)ws; ws += (size_t)M * HID * sizeof(_Float16);
  float*   fbv = (float*)ws;    ws += (size_t)RANK * DIM * sizeof(float);

  // 1) fbv = fast_b @ v
  k_fbv<<<(RANK * DIM) / 256, 256, 0, stream>>>(fast_b, v, fbv);
  // 2) W1 = tern(w_fc)*s + gate * u @ fbv  (folded adapter), f16 N-major
  k_prep_w<DIM, true><<<HID, 256, 0, stream>>>(w_fc, u, fbv, gate, W1);
  // 3) W2 = tern(w_proj)*s, f16 N-major
  k_prep_w<HID, false><<<DIM, 256, 0, stream>>>(w_proj, nullptr, nullptr, nullptr, W2);
  // 4) x -> f16
  k_cvt<<<(M * DIM / 8 + 255) / 256, 256, 0, stream>>>(x, xh, M * DIM);
  // 5) GEMM1 + relu^2 epilogue -> f16 intermediate
  dim3 g1(M / 128, HID / 128);
  k_gemm<HID, DIM, true><<<g1, 256, 0, stream>>>(xh, W1, C1, nullptr, M);
  // 6) GEMM2 -> f32 output
  dim3 g2(M / 128, DIM / 128);
  k_gemm<DIM, HID, false><<<g2, 256, 0, stream>>>(C1, W2, nullptr, out, M);
}
